// MAGCRN_79121887527513
// MI455X (gfx1250) — matrix operand order repaired and checked by reading, not compiled, not run on hardware
//
#include <hip/hip_runtime.h>
#include <hip/hip_bf16.h>
#include <math.h>
#include <stdint.h>

// ---------------------------------------------------------------------------
// CDNA5 (gfx1250) WMMA types.  wave32; 16x16x32 bf16 WMMA, f32 accumulate.
// ---------------------------------------------------------------------------
typedef __attribute__((ext_vector_type(16))) __bf16 v16bf;
typedef __attribute__((ext_vector_type(8)))  float  v8f;
typedef __attribute__((ext_vector_type(4)))  unsigned int v4u;
typedef __attribute__((ext_vector_type(8)))  int  v8i;
typedef __attribute__((ext_vector_type(4)))  int  v4i;

// Tensor Data Mover path: ROCm 7.2 exposes the 5-arg builtin; the
// amdgpu-toolchain lane ships the TDM header and uses a 6-arg form.
// Use TDM only on the 5-arg toolchain; otherwise fall back to VALU staging.
#if defined(__gfx1250__) && __has_builtin(__builtin_amdgcn_tensor_load_to_lds) && \
    __has_builtin(__builtin_amdgcn_s_wait_tensorcnt) && \
    !__has_include(<hip/amd_detail/amd_gfx1250_TDM.h>)
#define USE_TDM_5ARG 1
#endif

union FragAB { v16bf v; unsigned u[8]; };
union FragC  { v8f   v; float    f[8]; };

__device__ __forceinline__ unsigned short f2bf(float f) {
    unsigned u = __float_as_uint(f);
    u += 0x7fffu + ((u >> 16) & 1u);          // round-to-nearest-even
    return (unsigned short)(u >> 16);
}
__device__ __forceinline__ float bf2f(unsigned short h) {
    return __uint_as_float(((unsigned)h) << 16);
}
__device__ __forceinline__ float sigmoidf_(float x) { return 1.0f / (1.0f + __expf(-x)); }

// ---------------------------------------------------------------------------
// Generic batched GEMM: C[z] = A[z] (MxK, bf16) * B[z] (KxN, bf16), f32 acc.
// Block tile 64x64, 4 waves, each wave owns a 32x32 patch (2x2 WMMA tiles).
// A tile arrives via the Tensor Data Mover (async DMA into LDS, TENSORcnt
// gated) when available; B is staged transposed by the waves.  Fragments are
// packed per the CDNA5 16-bit A/B VGPR layouts (ISA 7.12.2).
// ---------------------------------------------------------------------------
__global__ void __launch_bounds__(128)
gemm_bf16_wmma(const unsigned short* __restrict__ A,
               const unsigned short* __restrict__ B,
               float* __restrict__ C32, unsigned short* __restrict__ Cbf,
               int M, int N, int K, int lda, int ldb, int ldc,
               long long sAz, long long sBz, long long sCz)
{
    __shared__ unsigned short sA [64][32];   // [m][k]
    __shared__ unsigned short sBt[64][32];   // [n][k]  (B transposed on the way in)

    const unsigned short* Ab = A + (long long)blockIdx.z * sAz;
    const unsigned short* Bb = B + (long long)blockIdx.z * sBz;
    const long long coff = (long long)blockIdx.z * sCz;

    const int row0 = blockIdx.y * 64;
    const int col0 = blockIdx.x * 64;
    const int tid  = threadIdx.x;
    const int lane = tid & 31;
    const int wave = tid >> 5;
    const int half = lane >> 4;
    const int l16  = lane & 15;
    const int wm   = (wave >> 1) * 32;
    const int wn   = (wave & 1) * 32;

    FragC acc[2][2];
#pragma unroll
    for (int a = 0; a < 2; ++a)
#pragma unroll
        for (int b2 = 0; b2 < 2; ++b2)
#pragma unroll
            for (int j = 0; j < 8; ++j) acc[a][b2].f[j] = 0.f;

    for (int kk = 0; kk < K; kk += 32) {
#if defined(USE_TDM_5ARG)
        // ---- Async A tile (64x32 bf16) via Tensor Data Mover ----
        if (tid == 0) {
            unsigned long long ga = (unsigned long long)(uintptr_t)Ab +
                                    ((unsigned long long)row0 * (unsigned)lda + (unsigned)kk) * 2ull;
            unsigned ldsa = (unsigned)(uintptr_t)&sA[0][0];   // low 32b of LDS aperture = offset
            unsigned dim0 = (unsigned)(K - kk);               // remaining K (OOB -> zero fill)
            unsigned dim1 = (unsigned)(M - row0);             // remaining M rows
            v4u g0;
            g0.x = 1u;                                        // count=1, user descriptor
            g0.y = ldsa;                                      // lds_addr
            g0.z = (unsigned)ga;                              // global_addr[31:0]
            g0.w = (unsigned)((ga >> 32) & 0x01ffffffull) | (2u << 30);  // addr[56:32] | type=2
            v8i g1;
            g1[0] = (int)(1u << 16);                          // data_size = 2 bytes
            g1[1] = (int)((dim0 & 0xffffu) << 16);            // tensor_dim0[15:0]
            g1[2] = (int)((dim0 >> 16) | ((dim1 & 0xffffu) << 16)); // dim0 hi | dim1 lo
            g1[3] = (int)((dim1 >> 16) | (32u << 16));        // dim1 hi | tile_dim0 = 32
            g1[4] = 64;                                       // tile_dim1 = 64 (tile_dim2 = 0)
            g1[5] = lda;                                      // tensor_dim0_stride[31:0]
            g1[6] = 0;                                        // stride hi / dim1_stride lo
            g1[7] = 0;
            v4i zz = (v4i){0, 0, 0, 0};
            __builtin_amdgcn_tensor_load_to_lds(g0, g1, zz, zz, 0);
        }
#else
        // ---- A tile (64x32), packed-pair (uint) loads; zero-fill OOB ----
#pragma unroll 4
        for (int idx = tid; idx < 64 * 16; idx += 128) {
            int r = idx >> 4, c2 = idx & 15;
            int gr = row0 + r, gk = kk + c2 * 2;
            unsigned val = 0u;
            if (gr < M) {
                if (gk + 1 < K)      val = *(const unsigned*)(Ab + (long long)gr * lda + gk);
                else if (gk < K)     val = (unsigned)Ab[(long long)gr * lda + gk];
            }
            *(unsigned*)&sA[r][c2 * 2] = val;
        }
#endif
        // ---- B tile transposed (read coalesced along n, write [n][k]) ----
#pragma unroll 4
        for (int idx = tid; idx < 32 * 64; idx += 128) {
            int kl = idx >> 6, nl = idx & 63;
            int gk = kk + kl, gn = col0 + nl;
            unsigned short v = 0;
            if (gk < K && gn < N) v = Bb[(long long)gk * ldb + gn];
            sBt[nl][kl] = v;
        }
#if defined(USE_TDM_5ARG)
        __builtin_amdgcn_s_wait_tensorcnt(0);
#endif
        __syncthreads();

        FragAB fa[2], fb[2];
#pragma unroll
        for (int mt = 0; mt < 2; ++mt) {
            int r = wm + mt * 16 + l16;
#pragma unroll
            for (int j = 0; j < 8; ++j) {
                int kb = ((j >= 4) ? 16 : 0) + half * 8 + (j & 3) * 2;   // A layout
                fa[mt].u[j] = *(const unsigned*)&sA[r][kb];
            }
        }
#pragma unroll
        for (int nt = 0; nt < 2; ++nt) {
            int nc = wn + nt * 16 + l16;
#pragma unroll
            for (int j = 0; j < 8; ++j) {
                int kb = ((j >= 4) ? 16 : 0) + half * 8 + (j & 3) * 2;   // B layout
                fb[nt].u[j] = *(const unsigned*)&sBt[nc][kb];
            }
        }
#pragma unroll
        for (int mt = 0; mt < 2; ++mt)
#pragma unroll
            for (int nt = 0; nt < 2; ++nt)
                acc[mt][nt].v = __builtin_amdgcn_wmma_f32_16x16x32_bf16(
                    false, fa[mt].v, false, fb[nt].v,
                    (short)0, acc[mt][nt].v, false, false);

#if !defined(USE_TDM_5ARG)
        // Prefetch next A tile slab (-> global_prefetch_b8).
        if (kk + 32 < K) {
            int pr = row0 + (tid & 63);
            if (pr < M) __builtin_prefetch(Ab + (long long)pr * lda + kk + 32, 0, 0);
        }
#else
        // Prefetch next B slab into GL2 while the TDM handles A.
        if (kk + 32 < K) {
            int pn = col0 + (tid & 63);
            if (pn < N) __builtin_prefetch(Bb + (long long)(kk + 32) * ldb + pn, 0, 0);
        }
#endif
        __syncthreads();
    }

    // Store (C/D layout: lane = N, vgpr j = M row within 8-row half).
#pragma unroll
    for (int mt = 0; mt < 2; ++mt)
#pragma unroll
        for (int nt = 0; nt < 2; ++nt) {
            int gcol = col0 + wn + nt * 16 + l16;
            if (gcol < N) {
#pragma unroll
                for (int j = 0; j < 8; ++j) {
                    int grow = row0 + wm + mt * 16 + half * 8 + j;
                    if (grow < M) {
                        long long o = coff + (long long)grow * ldc + gcol;
                        float vv = acc[mt][nt].f[j];
                        if (C32) C32[o] = vv;
                        if (Cbf) Cbf[o] = f2bf(vv);
                    }
                }
            }
        }
}

// ---------------------------------------------------------------------------
// adj = softmax(relu(E E^T), axis=1)  -> bf16.  One block per row, 256 thr.
// ---------------------------------------------------------------------------
__global__ void __launch_bounds__(256)
compute_adj(const float* __restrict__ emb, unsigned short* __restrict__ adj)
{
    __shared__ float srow[16];
    __shared__ float red[256];
    int i = blockIdx.x, tid = threadIdx.x;
    if (tid < 16) srow[tid] = emb[i * 16 + tid];
    __syncthreads();
    float v0 = 0.f, v1 = 0.f;
    {
        int j = tid; float s = 0.f;
        for (int e = 0; e < 16; ++e) s += srow[e] * emb[j * 16 + e];
        v0 = fmaxf(s, 0.f);
    }
    {
        int j = tid + 256; float s = 0.f;
        for (int e = 0; e < 16; ++e) s += srow[e] * emb[j * 16 + e];
        v1 = fmaxf(s, 0.f);
    }
    red[tid] = fmaxf(v0, v1); __syncthreads();
    for (int s = 128; s > 0; s >>= 1) { if (tid < s) red[tid] = fmaxf(red[tid], red[tid + s]); __syncthreads(); }
    float mx = red[0]; __syncthreads();
    v0 = __expf(v0 - mx); v1 = __expf(v1 - mx);
    red[tid] = v0 + v1; __syncthreads();
    for (int s = 128; s > 0; s >>= 1) { if (tid < s) red[tid] += red[tid + s]; __syncthreads(); }
    float inv = 1.0f / red[0];
    adj[(long long)i * 512 + tid]       = f2bf(v0 * inv);
    adj[(long long)i * 512 + tid + 256] = f2bf(v1 * inv);
}

// out[n,j] = sum_e emb[n,e]*pool[e,j]   (K=16); bf16 or f32 destination.
__global__ void gen_weights(const float* __restrict__ emb, const float* __restrict__ pool,
                            unsigned short* __restrict__ obf, float* __restrict__ of32,
                            int J, long long total)
{
    for (long long i = blockIdx.x * (long long)blockDim.x + threadIdx.x; i < total;
         i += (long long)gridDim.x * blockDim.x) {
        int n = (int)(i / J);
        long long j = i % J;
        float s = 0.f;
#pragma unroll
        for (int e = 0; e < 16; ++e) s += emb[n * 16 + e] * pool[(long long)e * J + j];
        if (obf) obf[i] = f2bf(s); else of32[i] = s;
    }
}

__global__ void zero_f32(float* p, long long n)
{
    for (long long i = blockIdx.x * (long long)blockDim.x + threadIdx.x; i < n;
         i += (long long)gridDim.x * blockDim.x) p[i] = 0.f;
}

__global__ void f32_to_bf16k(const float* __restrict__ s, unsigned short* __restrict__ d, long long n)
{
    for (long long i = blockIdx.x * (long long)blockDim.x + threadIdx.x; i < n;
         i += (long long)gridDim.x * blockDim.x) d[i] = f2bf(s[i]);
}

// xs = concat(xt, state) -> bf16; also written into xg k=0 slice.
__global__ void build_xs(const float* __restrict__ xt, long long xt_bstride, int xt_nstride, int din,
                         const float* __restrict__ state,
                         unsigned short* __restrict__ xs, unsigned short* __restrict__ xg,
                         int c, int Kc, long long total)
{
    for (long long i = blockIdx.x * (long long)blockDim.x + threadIdx.x; i < total;
         i += (long long)gridDim.x * blockDim.x) {
        int ch = (int)(i % c);
        long long bn = i / c;
        int n = (int)(bn & 511);
        int b = (int)(bn >> 9);
        float v;
        if (ch < din) v = xt[(long long)b * xt_bstride + (long long)n * xt_nstride + ch];
        else          v = state[bn * 64 + (ch - din)];
        unsigned short bf = f2bf(v);
        xs[bn * c  + ch] = bf;
        xg[bn * Kc + ch] = bf;
    }
}

// z,r = sigmoid(zr + gb); keep r; build xs2 = concat(xt, z*state) (bf16 + xg2 k0).
__global__ void gate_act(const float* __restrict__ zr, const float* __restrict__ gb,
                         const float* __restrict__ xt, long long xt_bstride, int xt_nstride, int din,
                         const float* __restrict__ state, float* __restrict__ rbuf,
                         unsigned short* __restrict__ xs2, unsigned short* __restrict__ xg2,
                         int c, int Kc, long long total)
{
    for (long long i = blockIdx.x * (long long)blockDim.x + threadIdx.x; i < total;
         i += (long long)gridDim.x * blockDim.x) {
        int h = (int)(i & 63);
        long long bn = i >> 6;
        int n = (int)(bn & 511);
        int b = (int)(bn >> 9);
        float z = sigmoidf_(zr[bn * 128 + h]      + gb[n * 128 + h]);
        float r = sigmoidf_(zr[bn * 128 + 64 + h] + gb[n * 128 + 64 + h]);
        rbuf[i] = r;
        float zs = z * state[i];
        int ch = din + h;
        unsigned short bfv = f2bf(zs);
        xs2[bn * c  + ch] = bfv;
        xg2[bn * Kc + ch] = bfv;
        if (h < din) {
            float xv = xt[(long long)b * xt_bstride + (long long)n * xt_nstride + h];
            unsigned short bx = f2bf(xv);
            xs2[bn * c  + h] = bx;
            xg2[bn * Kc + h] = bx;
        }
    }
}

// h = r*state + (1-r)*tanh(hclin + ub); state <- h; hidden[t] <- h.
__global__ void gru_update(const float* __restrict__ hclin, const float* __restrict__ ub,
                           const float* __restrict__ rbuf, float* __restrict__ state,
                           float* __restrict__ hidden_t, long long h_bstride, long long total)
{
    for (long long i = blockIdx.x * (long long)blockDim.x + threadIdx.x; i < total;
         i += (long long)gridDim.x * blockDim.x) {
        int h = (int)(i & 63);
        long long bn = i >> 6;
        int n = (int)(bn & 511);
        int b = (int)(bn >> 9);
        float hc = tanhf(hclin[i] + ub[n * 64 + h]);
        float r = rbuf[i];
        float s = state[i];
        float hn = r * s + (1.f - r) * hc;
        state[i] = hn;
        hidden_t[(long long)b * h_bstride + (long long)n * 64 + h] = hn;
    }
}

// fs[n,t] = sum_f (hlin[n, t*4+f] + hyp_b[t*4+f])
__global__ void hyp_reduce(const float* __restrict__ hlin, const float* __restrict__ hypb,
                           float* __restrict__ fs)
{
    int i = blockIdx.x * blockDim.x + threadIdx.x;
    if (i < 512 * 12) {
        int t = i % 12, n = i / 12;
        float s = 0.f;
#pragma unroll
        for (int f = 0; f < 4; ++f) s += hlin[n * 48 + t * 4 + f] + hypb[t * 4 + f];
        fs[i] = s;
    }
}

// hgen[b,t,n,h] = hidden1[b,T-1,n,h] * fs[n,t]  (bf16, feeds the V projection GEMM)
__global__ void gen_h(const float* __restrict__ hidden1, const float* __restrict__ fs,
                      unsigned short* __restrict__ hgenb, long long total)
{
    for (long long i = blockIdx.x * (long long)blockDim.x + threadIdx.x; i < total;
         i += (long long)gridDim.x * blockDim.x) {
        int h = (int)(i & 63);
        long long r = i >> 6;
        int n = (int)(r & 511);
        long long bt = r >> 9;
        int t = (int)(bt % 12);
        int b = (int)(bt / 12);
        float v = hidden1[(((long long)b * 12 + 11) * 512 + n) * 64 + h] * fs[n * 12 + t];
        hgenb[i] = f2bf(v);
    }
}

__global__ void bias_act(float* __restrict__ x, const float* __restrict__ bias,
                         int Hc, int relu, long long total)
{
    for (long long i = blockIdx.x * (long long)blockDim.x + threadIdx.x; i < total;
         i += (long long)gridDim.x * blockDim.x) {
        float v = x[i] + bias[i % Hc];
        if (relu) v = fmaxf(v, 0.f);
        x[i] = v;
    }
}

// Tiny T=12 attention per (b,n): 2 heads of d=32.
__global__ void __launch_bounds__(64)
attn_kernel(const float* __restrict__ qin, const float* __restrict__ kin,
            const float* __restrict__ vin, float* __restrict__ o)
{
    __shared__ float sq[12][64], sk[12][64], sv[12][64];
    int bn = blockIdx.x;
    int n = bn & 511, b = bn >> 9;
    int tid = threadIdx.x;
    for (int t = 0; t < 12; ++t) {
        long long idx = (((long long)b * 12 + t) * 512 + n) * 64 + tid;
        sq[t][tid] = qin[idx]; sk[t][tid] = kin[idx]; sv[t][tid] = vin[idx];
    }
    __syncthreads();
    if (tid < 24) {
        int hd = tid / 12, tq = tid % 12, base = hd * 32;
        float sc[12]; float mx = -3.4e38f;
        for (int tk = 0; tk < 12; ++tk) {
            float s = 0.f;
#pragma unroll
            for (int d = 0; d < 32; ++d) s += sq[tq][base + d] * sk[tk][base + d];
            s *= 0.17677669529663687f;                 // 1/sqrt(32)
            sc[tk] = s; mx = fmaxf(mx, s);
        }
        float sum = 0.f;
        for (int tk = 0; tk < 12; ++tk) { sc[tk] = __expf(sc[tk] - mx); sum += sc[tk]; }
        float inv = 1.f / sum;
        for (int d = 0; d < 32; ++d) {
            float a = 0.f;
            for (int tk = 0; tk < 12; ++tk) a += sc[tk] * sv[tk][base + d];
            o[(((long long)b * 12 + tq) * 512 + n) * 64 + base + d] = a * inv;
        }
    }
}

// out = LN(x + res) * g + b ; optional bf16 copy.  One 64-thread block per row.
__global__ void __launch_bounds__(64)
ln_res_kernel(const float* __restrict__ x, const float* __restrict__ res,
              const float* __restrict__ g, const float* __restrict__ bta,
              float* __restrict__ out, unsigned short* __restrict__ obf)
{
    __shared__ float red[64];
    long long row = blockIdx.x;
    int t = threadIdx.x;
    long long ix = row * 64 + t;
    float v = x[ix] + res[ix];
    red[t] = v; __syncthreads();
    for (int s = 32; s > 0; s >>= 1) { if (t < s) red[t] += red[t + s]; __syncthreads(); }
    float mu = red[0] * (1.f / 64.f);
    __syncthreads();
    float d = v - mu;
    red[t] = d * d; __syncthreads();
    for (int s = 32; s > 0; s >>= 1) { if (t < s) red[t] += red[t + s]; __syncthreads(); }
    float var = red[0] * (1.f / 64.f);
    float y = d * rsqrtf(var + 1e-5f) * g[t] + bta[t];
    out[ix] = y;
    if (obf) obf[ix] = f2bf(y);
}

__global__ void fc_kernel(const float* __restrict__ x, const float* __restrict__ w,
                          const float* __restrict__ b, float* __restrict__ y, long long rows)
{
    for (long long i = blockIdx.x * (long long)blockDim.x + threadIdx.x; i < rows;
         i += (long long)gridDim.x * blockDim.x) {
        float s = b[0];
#pragma unroll
        for (int h = 0; h < 64; ++h) s += x[i * 64 + h] * w[h];
        y[i] = s;
    }
}

// ---------------------------------------------------------------------------
// Host side
// ---------------------------------------------------------------------------
static inline int nb(long long total, int threads)
{
    long long g = (total + threads - 1) / threads;
    if (g > 1048576LL) g = 1048576LL;
    if (g < 1) g = 1;
    return (int)g;
}

static void launch_gemm(hipStream_t st, const unsigned short* A, const unsigned short* B,
                        float* C32, unsigned short* Cb,
                        int M, int N, int K, int lda, int ldb, int ldc,
                        long long sAz, long long sBz, long long sCz, int Z)
{
    dim3 g((unsigned)((N + 63) / 64), (unsigned)((M + 63) / 64), (unsigned)Z);
    gemm_bf16_wmma<<<g, dim3(128), 0, st>>>(A, B, C32, Cb, M, N, K, lda, ldb, ldc, sAz, sBz, sCz);
}

extern "C" void kernel_launch(void* const* d_in, const int* in_sizes, int n_in,
                              void* d_out, int out_size, void* d_ws, size_t ws_size,
                              hipStream_t stream)
{
    (void)in_sizes; (void)n_in; (void)out_size; (void)ws_size;

    const float* source = (const float*)d_in[0];
    const float* emb    = (const float*)d_in[2];
    const float* gwp0   = (const float*)d_in[3];
    const float* gbp0   = (const float*)d_in[4];
    const float* uwp0   = (const float*)d_in[5];
    const float* ubp0   = (const float*)d_in[6];
    const float* gwp1   = (const float*)d_in[7];
    const float* gbp1   = (const float*)d_in[8];
    const float* uwp1   = (const float*)d_in[9];
    const float* ubp1   = (const float*)d_in[10];
    const float* hypW   = (const float*)d_in[11];
    const float* hypb   = (const float*)d_in[12];
    const float* Wq     = (const float*)d_in[13];
    const float* Wk     = (const float*)d_in[14];
    const float* Wv     = (const float*)d_in[15];
    const float* fW1    = (const float*)d_in[16];
    const float* fW2    = (const float*)d_in[17];
    const float* bq     = (const float*)d_in[18];
    const float* bk     = (const float*)d_in[19];
    const float* bv     = (const float*)d_in[20];
    const float* fb1    = (const float*)d_in[21];
    const float* fb2    = (const float*)d_in[22];
    const float* ln1g   = (const float*)d_in[23];
    const float* ln1b   = (const float*)d_in[24];
    const float* ln2g   = (const float*)d_in[25];
    const float* ln2b   = (const float*)d_in[26];
    const float* fcW    = (const float*)d_in[27];
    const float* fcb    = (const float*)d_in[28];

    const long long BTN  = 32LL * 12 * 512;
    const long long BTNH = BTN * 64;
    const long long BNH  = 32LL * 512 * 64;

    char* ws = (char*)d_ws;
    size_t off = 0;
    auto alloc = [&](size_t bytes) -> void* {
        void* p = ws + off;
        off = (off + bytes + 255) & ~((size_t)255);
        return p;
    };

    // ---- persistent region ----
    unsigned short* adjb  = (unsigned short*)alloc((size_t)512 * 512 * 2);
    unsigned short* gW0   = (unsigned short*)alloc((size_t)512 * 16896 * 2);
    float*          gb0   = (float*)alloc((size_t)512 * 128 * 4);
    unsigned short* uW0   = (unsigned short*)alloc((size_t)512 * 8448 * 2);
    float*          ub0   = (float*)alloc((size_t)512 * 64 * 4);
    unsigned short* gW1   = (unsigned short*)alloc((size_t)512 * 32768 * 2);
    float*          gb1   = (float*)alloc((size_t)512 * 128 * 4);
    unsigned short* uW1   = (unsigned short*)alloc((size_t)512 * 16384 * 2);
    float*          ub1   = (float*)alloc((size_t)512 * 64 * 4);
    unsigned short* hypWb = (unsigned short*)alloc((size_t)16384 * 48 * 2);
    float*          hlin  = (float*)alloc((size_t)512 * 48 * 4);
    float*          fs    = (float*)alloc((size_t)512 * 12 * 4);
    unsigned short* Wqb   = (unsigned short*)alloc(4096 * 2);
    unsigned short* Wkb   = (unsigned short*)alloc(4096 * 2);
    unsigned short* Wvb   = (unsigned short*)alloc(4096 * 2);
    unsigned short* fW1b  = (unsigned short*)alloc(4096 * 2);
    unsigned short* fW2b  = (unsigned short*)alloc(4096 * 2);
    float*          hidden1 = (float*)alloc((size_t)BTNH * 4);
    unsigned short* outb    = (unsigned short*)alloc((size_t)BTNH * 2);
    size_t post_base = off;

    // ---- encoder scratch region ----
    float*          hidden0 = (float*)alloc((size_t)BTNH * 4);
    float*          state   = (float*)alloc((size_t)BNH * 4);
    unsigned short* xs      = (unsigned short*)alloc((size_t)32 * 512 * 128 * 2);
    unsigned short* xg      = (unsigned short*)alloc((size_t)32 * 512 * 256 * 2);
    unsigned short* xs2     = (unsigned short*)alloc((size_t)32 * 512 * 128 * 2);
    unsigned short* xg2     = (unsigned short*)alloc((size_t)32 * 512 * 256 * 2);
    float*          zr      = (float*)alloc((size_t)32 * 512 * 128 * 4);
    float*          rbuf    = (float*)alloc((size_t)BNH * 4);
    float*          hclin   = (float*)alloc((size_t)BNH * 4);

    // ---------------- setup: adjacency, per-node weights, hypernet ----------
    compute_adj<<<512, 256, 0, stream>>>(emb, adjb);

    auto gw = [&](const float* pool, unsigned short* obf, float* of32, int J) {
        long long tot = (long long)512 * J;
        gen_weights<<<nb(tot, 256), 256, 0, stream>>>(emb, pool, obf, of32, J, tot);
    };
    gw(gwp0, gW0, nullptr, 16896);  gw(gbp0, nullptr, gb0, 128);
    gw(uwp0, uW0, nullptr, 8448);   gw(ubp0, nullptr, ub0, 64);
    gw(gwp1, gW1, nullptr, 32768);  gw(gbp1, nullptr, gb1, 128);
    gw(uwp1, uW1, nullptr, 16384);  gw(ubp1, nullptr, ub1, 64);

    f32_to_bf16k<<<nb(16384LL * 48, 256), 256, 0, stream>>>(hypW, hypWb, 16384LL * 48);
    // hlin = uW1[512,16384] @ hypW[16384,48]
    launch_gemm(stream, uW1, hypWb, hlin, nullptr, 512, 48, 16384,
                16384, 48, 48, 0, 0, 0, 1);
    hyp_reduce<<<nb(512 * 12, 256), 256, 0, stream>>>(hlin, hypb, fs);

    f32_to_bf16k<<<nb(4096, 256), 256, 0, stream>>>(Wq,  Wqb,  4096);
    f32_to_bf16k<<<nb(4096, 256), 256, 0, stream>>>(Wk,  Wkb,  4096);
    f32_to_bf16k<<<nb(4096, 256), 256, 0, stream>>>(Wv,  Wvb,  4096);
    f32_to_bf16k<<<nb(4096, 256), 256, 0, stream>>>(fW1, fW1b, 4096);
    f32_to_bf16k<<<nb(4096, 256), 256, 0, stream>>>(fW2, fW2b, 4096);

    // ---------------- GRU encoder: 2 layers x 12 timesteps ------------------
    for (int layer = 0; layer < 2; ++layer) {
        const int c   = (layer == 0) ? 66 : 128;
        const int Kc  = 2 * c;
        const int din = (layer == 0) ? 2 : 64;
        const float* xt0 = (layer == 0) ? source : hidden0;
        const long long bstride = (long long)12 * 512 * din;
        const int nstride = din;
        unsigned short* gWl = layer ? gW1 : gW0;
        float*          gbl = layer ? gb1 : gb0;
        unsigned short* uWl = layer ? uW1 : uW0;
        float*          ubl = layer ? ub1 : ub0;
        float*          hid = layer ? hidden1 : hidden0;

        zero_f32<<<nb(BNH, 256), 256, 0, stream>>>(state, BNH);

        for (int t = 0; t < 12; ++t) {
            const float* xt = xt0 + (long long)t * 512 * nstride;
            const long long totc = 32LL * 512 * c;

            build_xs<<<nb(totc, 256), 256, 0, stream>>>(
                xt, bstride, nstride, din, state, xs, xg, c, Kc, totc);

            // graph diffusion: xg[:, :, 1, :] = adj @ xs   (z = batch)
            launch_gemm(stream, adjb, xs, nullptr, xg + c,
                        512, c, 512, 512, c, Kc,
                        0, (long long)512 * c, (long long)512 * Kc, 32);

            // gate gconv: zr[b,n,:] = xg[b,n].flat @ gW[n]   (z = node)
            launch_gemm(stream, xg, gWl, zr, nullptr,
                        32, 128, Kc, 512 * Kc, 128, 512 * 128,
                        (long long)Kc, (long long)Kc * 128, 128, 512);

            gate_act<<<nb(BNH, 256), 256, 0, stream>>>(
                zr, gbl, xt, bstride, nstride, din, state, rbuf, xs2, xg2, c, Kc, BNH);

            launch_gemm(stream, adjb, xs2, nullptr, xg2 + c,
                        512, c, 512, 512, c, Kc,
                        0, (long long)512 * c, (long long)512 * Kc, 32);

            // update gconv: hclin[b,n,:] = xg2[b,n].flat @ uW[n]
            launch_gemm(stream, xg2, uWl, hclin, nullptr,
                        32, 64, Kc, 512 * Kc, 64, 512 * 64,
                        (long long)Kc, (long long)Kc * 64, 64, 512);

            gru_update<<<nb(BNH, 256), 256, 0, stream>>>(
                hclin, ubl, rbuf, state, hid + (long long)t * 512 * 64,
                (long long)12 * 512 * 64, BNH);
        }
    }

    // ---------------- attention + FFN head (reuse encoder scratch) ----------
    off = post_base;
    float* obuf = (float*)alloc((size_t)BTNH * 4);
    size_t stage2 = off;
    unsigned short* hgenb = (unsigned short*)alloc((size_t)BTNH * 2);
    float* qb  = (float*)alloc((size_t)BTNH * 4);
    float* kb2 = (float*)alloc((size_t)BTNH * 4);
    float* vb2 = (float*)alloc((size_t)BTNH * 4);

    f32_to_bf16k<<<nb(BTNH, 256), 256, 0, stream>>>(hidden1, outb, BTNH);
    gen_h<<<nb(BTNH, 256), 256, 0, stream>>>(hidden1, fs, hgenb, BTNH);

    launch_gemm(stream, outb,  Wqb, qb,  nullptr, (int)BTN, 64, 64, 64, 64, 64, 0, 0, 0, 1);
    bias_act<<<nb(BTNH, 256), 256, 0, stream>>>(qb,  bq, 64, 0, BTNH);
    launch_gemm(stream, outb,  Wkb, kb2, nullptr, (int)BTN, 64, 64, 64, 64, 64, 0, 0, 0, 1);
    bias_act<<<nb(BTNH, 256), 256, 0, stream>>>(kb2, bk, 64, 0, BTNH);
    launch_gemm(stream, hgenb, Wvb, vb2, nullptr, (int)BTN, 64, 64, 64, 64, 64, 0, 0, 0, 1);
    bias_act<<<nb(BTNH, 256), 256, 0, stream>>>(vb2, bv, 64, 0, BTNH);

    attn_kernel<<<32 * 512, 64, 0, stream>>>(qb, kb2, vb2, obuf);

    off = stage2;   // q/k/v/hgen dead
    float*          val  = (float*)alloc((size_t)BTNH * 4);
    unsigned short* valb = (unsigned short*)alloc((size_t)BTNH * 2);
    float*          t1   = (float*)alloc((size_t)BTNH * 4);
    unsigned short* t1b  = (unsigned short*)alloc((size_t)BTNH * 2);
    float*          ffo  = (float*)alloc((size_t)BTNH * 4);
    float*          outh = (float*)alloc((size_t)BTNH * 4);

    ln_res_kernel<<<(int)BTN, 64, 0, stream>>>(obuf, hidden1, ln1g, ln1b, val, valb);

    launch_gemm(stream, valb, fW1b, t1, nullptr, (int)BTN, 64, 64, 64, 64, 64, 0, 0, 0, 1);
    bias_act<<<nb(BTNH, 256), 256, 0, stream>>>(t1, fb1, 64, 1, BTNH);
    f32_to_bf16k<<<nb(BTNH, 256), 256, 0, stream>>>(t1, t1b, BTNH);
    launch_gemm(stream, t1b, fW2b, ffo, nullptr, (int)BTN, 64, 64, 64, 64, 64, 0, 0, 0, 1);
    bias_act<<<nb(BTNH, 256), 256, 0, stream>>>(ffo, fb2, 64, 0, BTNH);

    ln_res_kernel<<<(int)BTN, 64, 0, stream>>>(ffo, val, ln2g, ln2b, outh, nullptr);

    fc_kernel<<<nb(BTN, 256), 256, 0, stream>>>(outh, fcW, fcb, (float*)d_out, BTN);
}
